// MultiHeadAttention_89086211653937
// MI455X (gfx1250) — matrix operand order repaired and checked by reading, not compiled, not run on hardware
//
#include <hip/hip_runtime.h>
#include <hip/hip_bf16.h>

typedef __attribute__((ext_vector_type(16))) __bf16 v16bf;
typedef __attribute__((ext_vector_type(8)))  __bf16 v8bf;
typedef __attribute__((ext_vector_type(8)))  float  v8f;

#define WMMA_BF16(a, b, c) \
  __builtin_amdgcn_wmma_f32_16x16x32_bf16(false, (a), false, (b), (short)0, (c), false, false)

constexpr int B_ = 4, S_ = 1024, D_ = 1024, H_ = 16, HS = 64, D3 = 3072;

// contiguous 32B -> B fragment (32x16 bf16: lanes 0-15 K=0..15 col=lane, lanes 16-31 K=16..31)
__device__ inline v16bf ld_b16(const __bf16* p) { return *(const v16bf*)p; }

// A fragment (16x32 bf16): lane row = lane&15; contiguous K [16h..16h+15]
__device__ inline v16bf ld_a16(const __bf16* p) {
  v8bf lo = *(const v8bf*)p;
  v8bf hi = *(const v8bf*)(p + 8);
  v16bf a;
#pragma unroll
  for (int i = 0; i < 8; i++) { a[i] = lo[i]; a[i + 8] = hi[i]; }
  return a;
}

__device__ inline float redmax16(float x) {
#pragma unroll
  for (int d = 1; d < 16; d <<= 1) x = fmaxf(x, __shfl_xor(x, d, 32));
  return x;
}
__device__ inline float redsum16(float x) {
#pragma unroll
  for (int d = 1; d < 16; d <<= 1) x += __shfl_xor(x, d, 32);
  return x;
}

// ---------------- f32 -> bf16 conversion ----------------
__global__ void k_f32_to_bf16(const float* __restrict__ s, __bf16* __restrict__ d, int n) {
  int i = blockIdx.x * blockDim.x + threadIdx.x;
  if (i < n) d[i] = (__bf16)s[i];
}

// ---------------- transpose V: qkvb[b*S+s][2D + h*hs + d] -> vT[b][h][d][s] ----------------
__global__ void k_transpose_v(const __bf16* __restrict__ qkvb, __bf16* __restrict__ vT) {
  int i = blockIdx.x * blockDim.x + threadIdx.x;  // over B*H*HS*S
  int s = i & (S_ - 1);
  int t = i >> 10;
  int d = t & (HS - 1);
  t >>= 6;
  int h = t & (H_ - 1);
  int b = t >> 4;
  vT[i] = qkvb[(size_t)(b * S_ + s) * D3 + 2 * D_ + h * HS + d];
}

// ---------------- generic GEMM: Y[M][N] = A[M][K] @ B[N][K]^T (torch Linear) ----------------
// block = 128 threads = 4 waves; each wave computes a 32x64 C strip (B fragments reused 2x).
template <typename OutT>
__global__ __launch_bounds__(128) void k_gemm_nt(const __bf16* __restrict__ A,
                                                 const __bf16* __restrict__ Bm,
                                                 OutT* __restrict__ Y,
                                                 int M, int N, int K) {
  const int lane = threadIdx.x & 31;
  const int wave = threadIdx.x >> 5;
  const int r0 = blockIdx.x * 128 + wave * 32;
  const int c0 = blockIdx.y * 64;
  const int rl = lane & 15, hh = lane >> 4;
  v8f acc[2][4] = {};
  for (int kb = 0; kb < K; kb += 32) {
    const __bf16* pa0 = A + (size_t)(r0 + rl) * K + kb + 16 * hh;
    const __bf16* pa1 = pa0 + (size_t)16 * K;
    if (kb + 32 < K) {  // gfx1250 global_prefetch for next K chunk
      __builtin_prefetch((const void*)(pa0 + 32), 0, 0);
      __builtin_prefetch((const void*)(pa1 + 32), 0, 0);
    }
    v16bf af0 = ld_a16(pa0);
    v16bf af1 = ld_a16(pa1);
#pragma unroll
    for (int nt = 0; nt < 4; nt++) {
      const __bf16* pb = Bm + (size_t)(c0 + nt * 16 + rl) * K + kb + 16 * hh;
      v16bf bfrag = ld_b16(pb);
      acc[0][nt] = WMMA_BF16(af0, bfrag, acc[0][nt]);
      acc[1][nt] = WMMA_BF16(af1, bfrag, acc[1][nt]);
    }
  }
#pragma unroll
  for (int mt = 0; mt < 2; mt++)
#pragma unroll
    for (int nt = 0; nt < 4; nt++)
#pragma unroll
      for (int g = 0; g < 8; g++)
        Y[(size_t)(r0 + mt * 16 + g + 8 * hh) * N + c0 + nt * 16 + rl] = (OutT)acc[mt][nt][g];
}

// ---------------- fused flash attention with Transformer-XL rel-shift ----------------
// grid = (S/16, H, B), block = 32 (one wave). Single-wave WG: barriers are S_NOPs.
__global__ __launch_bounds__(32) void k_attn(const __bf16* __restrict__ qkvb,
                                             const __bf16* __restrict__ vT,
                                             const __bf16* __restrict__ relw,  // [S][D] bf16
                                             const float* __restrict__ uvec,
                                             const float* __restrict__ vvec,
                                             __bf16* __restrict__ attnCat) {
  __shared__ __bf16 p_lds[16 * 32];   // P staging for K=32 PV
  __shared__ float  rel_lds[16 * 48]; // 48-wide rel band for a 32-key pair
  const int lane = threadIdx.x & 31;
  const int rl = lane & 15, hh = lane >> 4;
  const int b = blockIdx.z, h = blockIdx.y, i0 = blockIdx.x * 16;

  // A fragments for (q+u) [content] and (q+v) [rel], K chunks {0..31, 32..63}
  v16bf au[2], av[2];
#pragma unroll
  for (int kk = 0; kk < 2; kk++) {
    const __bf16* qp = qkvb + (size_t)(b * S_ + i0 + rl) * D3 + h * HS + kk * 32 + 16 * hh;
    v8bf lo = *(const v8bf*)qp;
    v8bf hi = *(const v8bf*)(qp + 8);
    const float* up = uvec + h * HS + kk * 32 + 16 * hh;
    const float* vp = vvec + h * HS + kk * 32 + 16 * hh;
#pragma unroll
    for (int i = 0; i < 8; i++) {
      float ql = (float)lo[i], qh = (float)hi[i];
      au[kk][i] = (__bf16)(ql + up[i]);
      au[kk][i + 8] = (__bf16)(qh + up[i + 8]);
      av[kk][i] = (__bf16)(ql + vp[i]);
      av[kk][i + 8] = (__bf16)(qh + vp[i + 8]);
    }
  }

  const __bf16* kbase = qkvb + (size_t)b * S_ * D3 + D_ + h * HS;   // K rows
  const __bf16* vbase = vT + (size_t)(b * H_ + h) * HS * S_;        // vT rows (per hs-dim)
  const __bf16* rbase = relw + h * HS;

  float m_[8], l_[8];
  v8f o[4] = {};
#pragma unroll
  for (int g = 0; g < 8; g++) { m_[g] = -3.0e38f; l_[g] = 0.0f; }

  const int nKeys = i0 + 16;
  int j0 = 0;
  // ================= paired key tiles: 32 keys per iteration, full-K PV =================
  for (; j0 + 32 <= nKeys; j0 += 32) {
    v8f s0 = {}, s1 = {};
#pragma unroll
    for (int kk = 0; kk < 2; kk++) {
      const __bf16* kp0 = kbase + (size_t)(j0 + rl) * D3 + kk * 32 + 16 * hh;
      const __bf16* kp1 = kp0 + (size_t)16 * D3;
      s0 = WMMA_BF16(au[kk], ld_b16(kp0), s0);
      s1 = WMMA_BF16(au[kk], ld_b16(kp1), s1);
    }
    // rel band: cols c=0..47, t = tbase + c  (clamped reads only feed masked entries)
    const int tbase = j0 - i0 + S_ - 16;
    v8f pr[3] = {};
#pragma unroll
    for (int cf = 0; cf < 3; cf++) {
      int t = tbase + cf * 16 + rl;
      if (t > S_ - 1) t = S_ - 1;
      const __bf16* rp = rbase + (size_t)t * D_ + 16 * hh;
#pragma unroll
      for (int kk = 0; kk < 2; kk++)
        pr[cf] = WMMA_BF16(av[kk], ld_b16(rp + kk * 32), pr[cf]);
    }
#pragma unroll
    for (int cf = 0; cf < 3; cf++)
#pragma unroll
      for (int g = 0; g < 8; g++)
        rel_lds[(g + 8 * hh) * 48 + cf * 16 + rl] = pr[cf][g];
    __syncthreads();
    // combined online-softmax update for 32 keys
    float p0_[8], p1_[8];
#pragma unroll
    for (int g = 0; g < 8; g++) {
      int r = g + 8 * hh;
      float sh0 = rel_lds[r * 48 + (rl - r + 15)];   // tile0: c = n - r + 15
      float sh1 = rel_lds[r * 48 + (rl - r + 31)];   // tile1: c = n - r + 31
      float v0 = (s0[g] + sh0) * 0.125f;             // tile0 never masked (j0+15 < i0)
      float v1 = (s1[g] + sh1) * 0.125f;
      if (j0 + 16 + rl > i0 + r) v1 = -3.0e38f;      // causal mask (tile1 may touch diagonal)
      float mx = redmax16(fmaxf(v0, v1));
      float mnew = fmaxf(m_[g], mx);
      float alpha = __expf(m_[g] - mnew);
      float p0 = __expf(v0 - mnew);
      float p1 = __expf(v1 - mnew);
      l_[g] = l_[g] * alpha + redsum16(p0 + p1);
      m_[g] = mnew;
      p0_[g] = p0;
      p1_[g] = p1;
#pragma unroll
      for (int nt = 0; nt < 4; nt++) o[nt][g] *= alpha;
    }
    __syncthreads();  // previous iteration's A_p reads complete before overwrite
#pragma unroll
    for (int g = 0; g < 8; g++) {
      p_lds[(g + 8 * hh) * 32 + rl] = (__bf16)p0_[g];
      p_lds[(g + 8 * hh) * 32 + 16 + rl] = (__bf16)p1_[g];
    }
    __syncthreads();
    v16bf ap = ld_a16(p_lds + rl * 32 + 16 * hh);
    // PV, full K=32: v rows j0..j0+31 all valid here
#pragma unroll
    for (int nt = 0; nt < 4; nt++) {
      const __bf16* vp = vbase + (size_t)(nt * 16 + rl) * S_ + j0 + 16 * hh;
      o[nt] = WMMA_BF16(ap, ld_b16(vp), o[nt]);
    }
  }
  // ================= remainder tile (16 keys, zero-padded K half) =================
  if (j0 < nKeys) {
    v8f sc = {};
#pragma unroll
    for (int kk = 0; kk < 2; kk++) {
      const __bf16* kp = kbase + (size_t)(j0 + rl) * D3 + kk * 32 + 16 * hh;
      sc = WMMA_BF16(au[kk], ld_b16(kp), sc);
    }
    const int tbase = j0 - i0 + S_ - 16;
    v8f pr[2] = {};
#pragma unroll
    for (int cf = 0; cf < 2; cf++) {
      int t = tbase + cf * 16 + rl;
      if (t > S_ - 1) t = S_ - 1;
      const __bf16* rp = rbase + (size_t)t * D_ + 16 * hh;
#pragma unroll
      for (int kk = 0; kk < 2; kk++)
        pr[cf] = WMMA_BF16(av[kk], ld_b16(rp + kk * 32), pr[cf]);
    }
#pragma unroll
    for (int cf = 0; cf < 2; cf++)
#pragma unroll
      for (int g = 0; g < 8; g++)
        rel_lds[(g + 8 * hh) * 48 + cf * 16 + rl] = pr[cf][g];
    __syncthreads();
    float p_[8];
#pragma unroll
    for (int g = 0; g < 8; g++) {
      int r = g + 8 * hh;
      float sh = rel_lds[r * 48 + (rl - r + 15)];
      float val = (sc[g] + sh) * 0.125f;
      if (j0 + rl > i0 + r) val = -3.0e38f;
      float mx = redmax16(val);
      float mnew = fmaxf(m_[g], mx);
      float alpha = __expf(m_[g] - mnew);
      float p = __expf(val - mnew);
      l_[g] = l_[g] * alpha + redsum16(p);
      m_[g] = mnew;
      p_[g] = p;
#pragma unroll
      for (int nt = 0; nt < 4; nt++) o[nt][g] *= alpha;
    }
    __syncthreads();
#pragma unroll
    for (int g = 0; g < 8; g++) {
      p_lds[(g + 8 * hh) * 32 + rl] = (__bf16)p_[g];
      p_lds[(g + 8 * hh) * 32 + 16 + rl] = (__bf16)0.0f;  // zero K half (pairs dirtied it)
    }
    __syncthreads();
    v16bf ap = ld_a16(p_lds + rl * 32 + 16 * hh);
    int ks = j0 + 16 * hh;
    if (ks > S_ - 16) ks = S_ - 16;  // OOB K half multiplies zero P
#pragma unroll
    for (int nt = 0; nt < 4; nt++) {
      const __bf16* vp = vbase + (size_t)(nt * 16 + rl) * S_ + ks;
      o[nt] = WMMA_BF16(ap, ld_b16(vp), o[nt]);
    }
  }
  // ---- epilogue: O / l, write head-concat layout [B][S][D] ----
#pragma unroll
  for (int g = 0; g < 8; g++) {
    float inv = 1.0f / l_[g];
    int row = i0 + g + 8 * hh;
#pragma unroll
    for (int nt = 0; nt < 4; nt++)
      attnCat[(size_t)(b * S_ + row) * D_ + h * HS + nt * 16 + rl] = (__bf16)(o[nt][g] * inv);
  }
}

// ---------------- host-side orchestration ----------------
extern "C" void kernel_launch(void* const* d_in, const int* in_sizes, int n_in,
                              void* d_out, int out_size, void* d_ws, size_t ws_size,
                              hipStream_t stream) {
  (void)in_sizes; (void)n_in; (void)out_size; (void)ws_size;
  const float* x    = (const float*)d_in[0];
  const float* Wqkv = (const float*)d_in[1];
  const float* Wout = (const float*)d_in[2];
  const float* Wpos = (const float*)d_in[3];
  const float* u    = (const float*)d_in[4];
  const float* v    = (const float*)d_in[5];
  const float* rel  = (const float*)d_in[6];

  char* ws = (char*)d_ws;
  size_t off = 0;
  auto alloc = [&](size_t bytes) -> void* {
    void* p = ws + off;
    off += (bytes + 255) & ~(size_t)255;
    return p;
  };
  __bf16* xb    = (__bf16*)alloc((size_t)B_ * S_ * D_ * 2);       // 8 MB
  __bf16* Wqkvb = (__bf16*)alloc((size_t)D3 * D_ * 2);            // 6 MB
  __bf16* Woutb = (__bf16*)alloc((size_t)D_ * D_ * 2);            // 2 MB
  __bf16* Wposb = (__bf16*)alloc((size_t)D_ * D_ * 2);            // 2 MB
  __bf16* relb  = (__bf16*)alloc((size_t)S_ * D_ * 2);            // 2 MB
  __bf16* qkvb  = (__bf16*)alloc((size_t)B_ * S_ * D3 * 2);       // 24 MB
  __bf16* relwb = (__bf16*)alloc((size_t)S_ * D_ * 2);            // 2 MB
  __bf16* vTb   = (__bf16*)alloc((size_t)B_ * H_ * HS * S_ * 2);  // 8 MB
  __bf16* attnb = (__bf16*)alloc((size_t)B_ * S_ * D_ * 2);       // 8 MB

  const int T = 256;
  k_f32_to_bf16<<<(B_ * S_ * D_ + T - 1) / T, T, 0, stream>>>(x, xb, B_ * S_ * D_);
  k_f32_to_bf16<<<(D3 * D_ + T - 1) / T, T, 0, stream>>>(Wqkv, Wqkvb, D3 * D_);
  k_f32_to_bf16<<<(D_ * D_ + T - 1) / T, T, 0, stream>>>(Wout, Woutb, D_ * D_);
  k_f32_to_bf16<<<(D_ * D_ + T - 1) / T, T, 0, stream>>>(Wpos, Wposb, D_ * D_);
  k_f32_to_bf16<<<(S_ * D_ + T - 1) / T, T, 0, stream>>>(rel, relb, S_ * D_);

  // qkv = x @ Wqkv^T : [4096, 3072]
  k_gemm_nt<__bf16><<<dim3(B_ * S_ / 128, D3 / 64), 128, 0, stream>>>(xb, Wqkvb, qkvb,
                                                                      B_ * S_, D3, D_);
  // relw = rel_embed @ Wpos^T : [1024, 1024]
  k_gemm_nt<__bf16><<<dim3(S_ / 128, D_ / 64), 128, 0, stream>>>(relb, Wposb, relwb,
                                                                 S_, D_, D_);
  // vT[b][h][d][s]
  k_transpose_v<<<(B_ * H_ * HS * S_) / T, T, 0, stream>>>(qkvb, vTb);
  // fused flash attention (content + rel-shift + causal softmax + PV)
  k_attn<<<dim3(S_ / 16, H_, B_), 32, 0, stream>>>(qkvb, vTb, relwb, u, v, attnb);
  // out = attnCat @ Wout^T -> f32
  k_gemm_nt<float><<<dim3(B_ * S_ / 128, D_ / 64), 128, 0, stream>>>(attnb, Woutb,
                                                                     (float*)d_out,
                                                                     B_ * S_, D_, D_);
}
